// HybridRetention_18846316494925
// MI455X (gfx1250) — matrix operand order repaired and checked by reading, not compiled, not run on hardware
//
#include <hip/hip_runtime.h>
#include <hip/hip_bf16.h>
#include <math.h>

// ---------------------------------------------------------------------------
// HybridRetention on MI455X (gfx1250): persistent-grid recurrent scan.
//   B=4, S=2048, D=1024, H=16, hd=64
// 16 WGs x 128 threads (4 waves). Each wave owns one 16-col WMMA C-tile.
// Stage vectors are kept in bf16, pre-packed in WMMA A-fragment layout, and
// pulled into LDS with global_load_async_to_lds_b128 (ASYNCcnt) so the
// K-loop is: ds_load_b128 (A) + global_load_b128 (B) + v_wmma.
// ---------------------------------------------------------------------------

#define DIMD   1024
#define SEQ    2048
#define BSZ    4
#define NWG    16
#define TPB    128
#define NTILES 64          // D / 16 column tiles
#define KTILES 32          // D / 32 k tiles
#define STAGE_ELTS (KTILES * 32 * 16)   // 16384 bf16 = 32 KB per stage vector

typedef __attribute__((ext_vector_type(16))) __bf16 v16bf;
typedef __attribute__((ext_vector_type(8)))  float  v8f;
typedef __attribute__((ext_vector_type(8)))  int    v8i;

union BFrag { v8i i; v16bf h; };

// ---------------------------------------------------------------------------
// CDNA5 async global->LDS copy (ASYNCcnt) helpers.
// ---------------------------------------------------------------------------
__device__ __forceinline__ void async_cp16(unsigned lds_addr, const void* gaddr) {
  asm volatile("global_load_async_to_lds_b128 %0, %1, off"
               :: "v"(lds_addr), "v"(gaddr) : "memory");
}
__device__ __forceinline__ void wait_async0() {
  asm volatile("s_wait_asynccnt 0" ::: "memory");
}

// Cooperative copy of one 32KB stage vector (global, A-frag layout) into LDS.
__device__ __forceinline__ void stage_to_lds(__bf16* sdst, const __bf16* gsrc) {
  const char* g = (const char*)gsrc;
  unsigned ldsbase = (unsigned)(uintptr_t)sdst;   // low 32 bits = LDS address
#pragma unroll
  for (int i = 0; i < 16; ++i) {
    unsigned off = (unsigned)threadIdx.x * 256u + (unsigned)i * 16u;
    async_cp16(ldsbase + off, g + off);
  }
  wait_async0();
  __syncthreads();
}

// ---------------------------------------------------------------------------
// Grid barrier: monotonic counter in L2, agent-scope acq/rel.
// ---------------------------------------------------------------------------
__device__ __forceinline__ void grid_barrier(unsigned* ctr, unsigned nbar) {
  __builtin_amdgcn_fence(__ATOMIC_RELEASE, "agent");
  __syncthreads();
  if (threadIdx.x == 0) {
    __hip_atomic_fetch_add(ctr, 1u, __ATOMIC_RELAXED, __HIP_MEMORY_SCOPE_AGENT);
    while (__hip_atomic_load(ctr, __ATOMIC_RELAXED, __HIP_MEMORY_SCOPE_AGENT)
           < nbar * (unsigned)NWG) {
      __builtin_amdgcn_s_sleep(2);
    }
  }
  __syncthreads();
  __builtin_amdgcn_fence(__ATOMIC_ACQUIRE, "agent");
}

// ---------------------------------------------------------------------------
// Scatter one f32 value into the bf16 A-fragment-layout stage buffer.
// A layout (ISA 7.12.2, 16-bit A, 16x32 tile): lane<16: row=lane,
//   K=k0+{0..7,16..23}; lane>=16: row=lane-16, K=k0+{8..15,24..31}.
// For (row=b, k=col): lane = b + 16*((col>>3)&1),
//                     elem = (col&7) | ((col&16)>>1).
// ---------------------------------------------------------------------------
__device__ __forceinline__ void store_a_elem(__bf16* stage, int col, int b, float val) {
  int kt   = col >> 5, w = col & 31;
  int lane = b + ((w & 8) << 1);
  int e    = (w & 7) | ((w & 16) >> 1);
  stage[((kt * 32 + lane) << 4) + e] = (__bf16)val;
}

// One [16 x 16] C tile, K = 1024: 32 x v_wmma_f32_16x16x32_bf16.
// A from LDS (ds_load_b128), B from L2-resident packed weights.
__device__ __forceinline__ v8f gemm_tile(const __bf16* sA, const __bf16* Wp,
                                         int nt, v8f acc, int lane) {
#pragma unroll 4
  for (int kt = 0; kt < KTILES; ++kt) {
    BFrag a, b;
    a.i = *(const v8i*)(sA + ((kt * 32 + lane) << 4));
    b.i = *(const v8i*)(Wp + ((((size_t)kt * NTILES) + nt) * 32 + lane) * 16);
    acc = __builtin_amdgcn_wmma_f32_16x16x32_bf16(
        false, a.h, false, b.h, (short)0, acc, false, false);
  }
  return acc;
}

__device__ __forceinline__ v8f bias_init(float bval) {
  v8f a;
#pragma unroll
  for (int j = 0; j < 8; ++j) a[j] = bval;
  return a;
}

// ---------------------------------------------------------------------------
// Weight pre-pack: f32 [N][K] (used as X @ W.T) -> bf16 per-lane WMMA-B tiles.
// B layout (matches A): lanes 0-15: K={0..7,16..23} / N=lane; lanes 16-31: K={8..15,24..31}
// ---------------------------------------------------------------------------
__global__ void __launch_bounds__(256) pack_weight_bf16(const float* __restrict__ W,
                                                        __bf16* __restrict__ dst) {
  int idx = blockIdx.x * blockDim.x + threadIdx.x;
  if (idx >= DIMD * DIMD) return;
  int n = idx >> 10, kk = idx & 1023;
  int kt = kk >> 5, w = kk & 31;
  int ntl = n >> 4, ncol = n & 15;
  int lane = ncol + ((w & 8) << 1);
  int e    = (w & 7) | ((w & 16) >> 1);
  size_t off = ((((size_t)kt * NTILES) + ntl) * 32 + lane) * 16 + e;
  dst[off] = (__bf16)W[(size_t)n * DIMD + kk];
}

// Zero barrier counter + all four bf16 stage buffers (pad rows must be 0).
__global__ void __launch_bounds__(256) init_ws(unsigned* base, int nwords) {
  int i = blockIdx.x * blockDim.x + threadIdx.x;
  if (i < nwords) base[i] = 0u;
}

// ---------------------------------------------------------------------------
// Persistent recurrent kernel.
// ---------------------------------------------------------------------------
__global__ void __launch_bounds__(TPB) retention_scan(
    const float* __restrict__ q, const float* __restrict__ k,
    const float* __restrict__ v,
    const float* __restrict__ bi, const float* __restrict__ bg,
    const float* __restrict__ bo, const float* __restrict__ decay,
    const __bf16* __restrict__ Wi_p, const __bf16* __restrict__ Wg_p,
    const __bf16* __restrict__ A_p,  const __bf16* __restrict__ Bm_p,
    const __bf16* __restrict__ Wo_p,
    __bf16* stage_ret, __bf16* stage_inp, __bf16* stage_blend, __bf16* stage_state,
    unsigned* bar, float* __restrict__ out) {

  extern __shared__ __bf16 smem[];
  __bf16* sRET = smem;
  __bf16* sINP = smem + STAGE_ELTS;
  __bf16* sBLD = smem + 2 * STAGE_ELTS;
  __bf16* sSTA = smem + 3 * STAGE_ELTS;

  const int wg   = blockIdx.x;          // 0..15 -> columns [wg*64, wg*64+64)
  const int wave = threadIdx.x >> 5;    // 0..3
  const int lane = threadIdx.x & 31;
  const int nt   = wg * 4 + wave;       // global 16-col tile id
  const int col  = nt * 16 + (lane & 15);
  const bool elem = (lane < 16);        // lanes carrying batch rows 0..3

  float rr[BSZ]    = {0.f, 0.f, 0.f, 0.f};
  float st[BSZ]    = {0.f, 0.f, 0.f, 0.f};
  float inpReg[BSZ];
  const float dec   = decay[col >> 6];
  const float bival = bi[col];
  const float bgval = bg[col];
  const float boval = bo[col];

  unsigned nbar = 0;

  for (int t = 0; t < SEQ; ++t) {
    // ---- phase 0: r = dec*r + k*v ; retained = q*r (bf16 A-layout) -------
    if (elem) {
#pragma unroll
      for (int b = 0; b < BSZ; ++b) {
        size_t idx = ((size_t)b * SEQ + t) * DIMD + col;
        rr[b] = dec * rr[b] + k[idx] * v[idx];
        store_a_elem(stage_ret, col, b, q[idx] * rr[b]);
      }
      if (t + 1 < SEQ) {
        size_t nidx = ((size_t)t + 1) * DIMD + col;
        __builtin_prefetch(&q[nidx], 0, 1);   // global_prefetch_b8
        __builtin_prefetch(&k[nidx], 0, 1);
        __builtin_prefetch(&v[nidx], 0, 1);
      }
    }
    grid_barrier(bar, ++nbar);

    // ---- phase 1: inp = retained @ Wi.T + bi ----------------------------
    stage_to_lds(sRET, stage_ret);
    v8f acc = bias_init(bival);
    acc = gemm_tile(sRET, Wi_p, nt, acc, lane);
    if (elem) {
#pragma unroll
      for (int b = 0; b < BSZ; ++b) {
        inpReg[b] = acc[b];
        store_a_elem(stage_inp, col, b, acc[b]);
      }
    }
    grid_barrier(bar, ++nbar);

    // ---- phase 2: gate = sigmoid(inp @ Wg.T + bg); blended --------------
    stage_to_lds(sINP, stage_inp);
    acc = bias_init(bgval);
    acc = gemm_tile(sINP, Wg_p, nt, acc, lane);
    if (elem) {
#pragma unroll
      for (int b = 0; b < BSZ; ++b) {
        float g = 1.0f / (1.0f + __expf(-acc[b]));
        store_a_elem(stage_blend, col, b, g * st[b] + (1.0f - g) * inpReg[b]);
      }
    }
    grid_barrier(bar, ++nbar);

    // ---- phase 3: state = tanh(blended @ A.T + inp @ Bm.T) --------------
    stage_to_lds(sBLD, stage_blend);                 // sINP still resident
    acc = bias_init(0.0f);
    acc = gemm_tile(sBLD, A_p,  nt, acc, lane);
    acc = gemm_tile(sINP, Bm_p, nt, acc, lane);
    if (elem) {
#pragma unroll
      for (int b = 0; b < BSZ; ++b) {
        st[b] = tanhf(acc[b]);
        store_a_elem(stage_state, col, b, st[b]);
      }
    }
    grid_barrier(bar, ++nbar);

    // ---- phase 4: out = state @ Wo.T + bo -------------------------------
    stage_to_lds(sSTA, stage_state);
    acc = bias_init(boval);
    acc = gemm_tile(sSTA, Wo_p, nt, acc, lane);
    if (elem) {
#pragma unroll
      for (int b = 0; b < BSZ; ++b)
        out[((size_t)b * SEQ + t) * DIMD + col] = acc[b];
    }
    // no barrier: next phase-0 touches only stage_ret, whose readers all
    // passed this step's phase-1 barrier.
  }
}

// ---------------------------------------------------------------------------
extern "C" void kernel_launch(void* const* d_in, const int* in_sizes, int n_in,
                              void* d_out, int out_size, void* d_ws, size_t ws_size,
                              hipStream_t stream) {
  (void)in_sizes; (void)n_in; (void)out_size; (void)ws_size;
  const float* q     = (const float*)d_in[0];
  const float* k     = (const float*)d_in[1];
  const float* v     = (const float*)d_in[2];
  const float* Wi    = (const float*)d_in[3];
  const float* bi    = (const float*)d_in[4];
  const float* Wg    = (const float*)d_in[5];
  const float* bg    = (const float*)d_in[6];
  const float* A     = (const float*)d_in[7];
  const float* Bm    = (const float*)d_in[8];
  const float* Wo    = (const float*)d_in[9];
  const float* bo    = (const float*)d_in[10];
  const float* decay = (const float*)d_in[11];
  float* out = (float*)d_out;

  // workspace layout:
  //   [0,256)            barrier counter
  //   [256, 256+128K)    4 bf16 stage vectors (A-frag layout, 32KB each)
  //   [256+128K, +10MB)  5 packed bf16 weight matrices
  char* ws = (char*)d_ws;
  unsigned* bar       = (unsigned*)ws;
  __bf16* stage_base  = (__bf16*)(ws + 256);
  __bf16* stage_ret   = stage_base;
  __bf16* stage_inp   = stage_base + STAGE_ELTS;
  __bf16* stage_blend = stage_base + 2 * STAGE_ELTS;
  __bf16* stage_state = stage_base + 3 * STAGE_ELTS;
  __bf16* packbase    = (__bf16*)(ws + 256 + 4 * STAGE_ELTS * sizeof(__bf16));
  const size_t WSZ = (size_t)DIMD * DIMD;
  __bf16* Wi_p = packbase + 0 * WSZ;
  __bf16* Wg_p = packbase + 1 * WSZ;
  __bf16* A_p  = packbase + 2 * WSZ;
  __bf16* Bm_p = packbase + 3 * WSZ;
  __bf16* Wo_p = packbase + 4 * WSZ;

  // zero counter + stage buffers (pad rows must stay zero; ws is poisoned)
  const int zwords = (256 + 4 * STAGE_ELTS * (int)sizeof(__bf16)) / 4;
  init_ws<<<(zwords + 255) / 256, 256, 0, stream>>>(bar, zwords);

  const int pk_blocks = (DIMD * DIMD + 255) / 256;
  pack_weight_bf16<<<pk_blocks, 256, 0, stream>>>(Wi, Wi_p);
  pack_weight_bf16<<<pk_blocks, 256, 0, stream>>>(Wg, Wg_p);
  pack_weight_bf16<<<pk_blocks, 256, 0, stream>>>(A,  A_p);
  pack_weight_bf16<<<pk_blocks, 256, 0, stream>>>(Bm, Bm_p);
  pack_weight_bf16<<<pk_blocks, 256, 0, stream>>>(Wo, Wo_p);

  retention_scan<<<NWG, TPB, 4 * STAGE_ELTS * sizeof(__bf16), stream>>>(
      q, k, v, bi, bg, bo, decay,
      Wi_p, Wg_p, A_p, Bm_p, Wo_p,
      stage_ret, stage_inp, stage_blend, stage_state,
      bar, out);
}
